// DSSM_80135499809069
// MI455X (gfx1250) — matrix operand order repaired and checked by reading, not compiled, not run on hardware
//
#include <hip/hip_runtime.h>
#include <hip/hip_bf16.h>

typedef __attribute__((ext_vector_type(16))) __bf16 v16bf;
typedef __attribute__((ext_vector_type(8)))  float  v8f;
typedef __attribute__((ext_vector_type(4)))  float  f32x4;

#define BATCH 4
#define HH 48
#define WW 48
#define LL (HH*WW)          // 2304
#define DM 96
#define DI 192
#define NS 16
#define DTR 6
#define KD 3
#define TOK (BATCH*LL)      // 9216
#define CC (DTR + 2*NS)     // 38
#define CPAD 48             // x_proj N padded to 3 tiles
#define DTPAD 32            // dt rank padded to one 32-K chunk

// ---- WMMA 16x16x32 bf16 fragment loaders (CDNA5 ISA 7.12.2) ----
// A (16-bit 16x32): lane m = lane&15, half = lane>>4.
//   element j=0..7  <-> K = kbase + 8*half + j        (contiguous run)
//   element j=8..15 <-> K = kbase + 16 + 8*half + (j-8) (contiguous run)
// All sources padded so loads are unconditional float4 vectors.
__device__ __forceinline__ v16bf fragA_f32(const float* __restrict__ row, int half, int kbase) {
  const float* r0 = row + kbase + half * 8;
  const float* r1 = r0 + 16;
  f32x4 a0 = *(const f32x4*)(r0);
  f32x4 a1 = *(const f32x4*)(r0 + 4);
  f32x4 a2 = *(const f32x4*)(r1);
  f32x4 a3 = *(const f32x4*)(r1 + 4);
  v16bf f;
#pragma unroll
  for (int i = 0; i < 4; ++i) {
    f[i]      = (__bf16)a0[i];
    f[4 + i]  = (__bf16)a1[i];
    f[8 + i]  = (__bf16)a2[i];
    f[12 + i] = (__bf16)a3[i];
  }
  return f;
}

// B (16-bit 32x16): lane n = lane&15; same K order as A (j<8: kbase+8*half+j; j>=8: kbase+16+8*half+(j-8))
__device__ __forceinline__ v16bf fragB_f32(const float* __restrict__ row, int half, int kbase) {
  const float* r0 = row + kbase + half * 8;
  const float* r1 = r0 + 16;
  f32x4 b0 = *(const f32x4*)(r0);
  f32x4 b1 = *(const f32x4*)(r0 + 4);
  f32x4 b2 = *(const f32x4*)(r1);
  f32x4 b3 = *(const f32x4*)(r1 + 4);
  v16bf f;
#pragma unroll
  for (int i = 0; i < 4; ++i) {
    f[i]      = (__bf16)b0[i];
    f[4 + i]  = (__bf16)b1[i];
    f[8 + i]  = (__bf16)b2[i];
    f[12 + i] = (__bf16)b3[i];
  }
  return f;
}

__device__ __forceinline__ float silu_f(float v) { return v / (1.0f + __expf(-v)); }

// ---- Prep: pad x_proj_w (K,38,192) -> (K,48,192) with zero rows ----
__global__ void k_pack_xpw(const float* __restrict__ xpw, float* __restrict__ xpw_pad) {
  int idx = blockIdx.x * blockDim.x + threadIdx.x;
  if (idx >= KD * CPAD * DI) return;
  int d = idx % DI; int tmp = idx / DI; int c = tmp % CPAD; int k = tmp / CPAD;
  xpw_pad[idx] = (c < CC) ? xpw[((size_t)k * CC + c) * DI + d] : 0.0f;
}

// ---- Prep: pad dt_projs_w (K,192,6) -> (K,192,32) with zero cols ----
__global__ void k_pack_dtw(const float* __restrict__ dtw, float* __restrict__ dtw_pad) {
  int idx = blockIdx.x * blockDim.x + threadIdx.x;
  if (idx >= KD * DI * DTPAD) return;
  int r = idx % DTPAD; int tmp = idx / DTPAD; int d = tmp % DI; int k = tmp / DI;
  dtw_pad[idx] = (r < DTR) ? dtw[((size_t)k * DI + d) * DTR + r] : 0.0f;
}

// ---- Kernel 1: in_proj GEMM (T x 96) @ (96 x 384)^T. One wave = 1 Mtile x 4 Ntiles. ----
__global__ void k_inproj(const float* __restrict__ x, const float* __restrict__ w,
                         float* __restrict__ xcbuf, float* __restrict__ zbuf) {
  int lane = threadIdx.x, half = lane >> 4, ln = lane & 15;
  int mt = blockIdx.x, ng = blockIdx.y;                 // ng: group of 4 N tiles
  const float* arow = x + (size_t)(mt * 16 + ln) * DM;
  const float* brow0 = w + (size_t)((ng * 4) * 16 + ln) * DM;
  v8f acc[4] = {};
#pragma unroll
  for (int kb = 0; kb < DM; kb += 32) {
    v16bf a = fragA_f32(arow, half, kb);
#pragma unroll
    for (int i = 0; i < 4; ++i) {
      v16bf b = fragB_f32(brow0 + (size_t)i * 16 * DM, half, kb);
      acc[i] = __builtin_amdgcn_wmma_f32_16x16x32_bf16(false, a, false, b, (short)0, acc[i], false, false);
    }
  }
#pragma unroll
  for (int i = 0; i < 4; ++i) {
    int n = (ng * 4 + i) * 16 + ln;
#pragma unroll
    for (int r = 0; r < 8; ++r) {
      int t = mt * 16 + half * 8 + r;
      int b_ = t / LL, p = t % LL;
      float v = acc[i][r];
      if (n < DI) xcbuf[((size_t)(b_ * DI + n)) * LL + p] = v;          // (B,D,H,W)
      else        zbuf[(size_t)t * DI + (n - DI)] = silu_f(v);          // (T,D)
    }
  }
}

// ---- Kernel 2: depthwise 3x3 SAME conv + bias + SiLU -> token-major (T,D) ----
__global__ void k_conv(const float* __restrict__ xcbuf, const float* __restrict__ cw,
                       const float* __restrict__ cb, float* __restrict__ xconv_t) {
  int idx = blockIdx.x * blockDim.x + threadIdx.x;
  if (idx >= BATCH * DI * LL) return;
  int p = idx % LL; int tmp = idx / LL; int d = tmp % DI; int b = tmp / DI;
  int h = p / WW, w = p % WW;
  const float* src = xcbuf + ((size_t)(b * DI + d)) * LL;
  float acc = cb[d];
#pragma unroll
  for (int kh = 0; kh < 3; ++kh) {
    int hh = h + kh - 1;
    if (hh < 0 || hh >= HH) continue;
#pragma unroll
    for (int kw = 0; kw < 3; ++kw) {
      int ww2 = w + kw - 1;
      if (ww2 < 0 || ww2 >= WW) continue;
      acc = fmaf(src[hh * WW + ww2], cw[d * 9 + kh * 3 + kw], acc);
    }
  }
  xconv_t[((size_t)(b * LL + p)) * DI + d] = silu_f(acc);
}

// ---- Kernel 3: x_proj per k: (T x 192) @ (192 x 48pad)^T. One wave covers all 3 Ntiles.
// Writes Bs/Cs rows (stride 38) and zero-padded dt-rank rows (stride 32). ----
__global__ void k_xproj(const float* __restrict__ xconv_t, const float* __restrict__ xpw_pad,
                        float* __restrict__ xdbl, float* __restrict__ dtspad) {
  int lane = threadIdx.x, half = lane >> 4, ln = lane & 15;
  int mt = blockIdx.x, kk = blockIdx.y;
  const float* arow = xconv_t + (size_t)(mt * 16 + ln) * DI;
  const float* bbase = xpw_pad + ((size_t)kk * CPAD + ln) * DI;
  v8f acc[3] = {};
#pragma unroll
  for (int kb = 0; kb < DI; kb += 32) {
    v16bf a = fragA_f32(arow, half, kb);
#pragma unroll
    for (int i = 0; i < 3; ++i) {
      v16bf b = fragB_f32(bbase + (size_t)i * 16 * DI, half, kb);
      acc[i] = __builtin_amdgcn_wmma_f32_16x16x32_bf16(false, a, false, b, (short)0, acc[i], false, false);
    }
  }
#pragma unroll
  for (int i = 0; i < 3; ++i) {
    int cidx = i * 16 + ln;
#pragma unroll
    for (int r = 0; r < 8; ++r) {
      int t = mt * 16 + half * 8 + r;
      int b_ = t / LL, p = t % LL;
      size_t tok = ((size_t)(b_ * KD + kk)) * LL + p;
      float v = acc[i][r];
      if (cidx < CC)    xdbl[tok * CC + cidx] = v;                       // dt|Bs|Cs packed
      if (cidx < DTPAD) dtspad[tok * DTPAD + cidx] = (cidx < DTR) ? v : 0.0f;
    }
  }
}

// ---- Kernel 4: dt projection (padded rank 32 -> 192) + bias + softplus -> delta.
// One wave = 1 Mtile x 4 Ntiles, single K chunk. ----
__global__ void k_dtproj(const float* __restrict__ dtspad, const float* __restrict__ dtw_pad,
                         const float* __restrict__ dtb, float* __restrict__ delta) {
  int lane = threadIdx.x, half = lane >> 4, ln = lane & 15;
  int mt = blockIdx.x, ng = blockIdx.y, kk = blockIdx.z;   // ng: 0..2 -> 4 N tiles each
  int t0 = mt * 16 + ln;
  int b0 = t0 / LL, p0 = t0 % LL;
  const float* arow = dtspad + (((size_t)(b0 * KD + kk)) * LL + p0) * DTPAD;
  const float* bbase = dtw_pad + ((size_t)kk * DI + (ng * 4) * 16 + ln) * DTPAD;
  v8f acc[4] = {};
  v16bf a = fragA_f32(arow, half, 0);
#pragma unroll
  for (int i = 0; i < 4; ++i) {
    v16bf b = fragB_f32(bbase + (size_t)i * 16 * DTPAD, half, 0);
    acc[i] = __builtin_amdgcn_wmma_f32_16x16x32_bf16(false, a, false, b, (short)0, acc[i], false, false);
  }
#pragma unroll
  for (int i = 0; i < 4; ++i) {
    int d = (ng * 4 + i) * 16 + ln;
    float bias = dtb[kk * DI + d];
#pragma unroll
    for (int r = 0; r < 8; ++r) {
      int t = mt * 16 + half * 8 + r;
      int b_ = t / LL, p = t % LL;
      float v = acc[i][r] + bias;
      float sp = (v > 20.0f) ? v : log1pf(__expf(v));
      delta[(((size_t)(b_ * KD + kk)) * LL + p) * DI + d] = sp;
    }
  }
}

// ---- Kernel 5: selective scan. One block per (b,k): 192 threads (one per d, 6 waves).
// Direction folded into traversal order; output stored at original spatial position. ----
__global__ void k_scan(const float* __restrict__ xconv_t, const float* __restrict__ xdbl,
                       const float* __restrict__ delta_buf, const float* __restrict__ A_logs,
                       const float* __restrict__ Ds, float* __restrict__ ybuf) {
  int b = blockIdx.x / KD, k = blockIdx.x % KD;
  int d = threadIdx.x;
  __shared__ float sh[2 * NS];   // Bs[16] | Cs[16] for the current step
  float Ad[NS];
#pragma unroll
  for (int n = 0; n < NS; ++n)
    Ad[n] = -__expf(A_logs[((size_t)(k * DI + d)) * NS + n]);
  float h[NS];
#pragma unroll
  for (int n = 0; n < NS; ++n) h[n] = 0.0f;
  float Dsv = Ds[k * DI + d];

  for (int s = 0; s < LL; ++s) {
    int p = (k == 0) ? s : (k == 1) ? ((s % HH) * WW + (s / HH)) : (LL - 1 - s);
    size_t tok = ((size_t)(b * KD + k)) * LL + p;
    if (threadIdx.x < 2 * NS)
      sh[threadIdx.x] = xdbl[tok * CC + DTR + threadIdx.x];
    __syncthreads();

    if (s + 1 < LL) {  // prefetch next step's per-lane operands
      int pn = (k == 0) ? (s + 1) : (k == 1) ? (((s + 1) % HH) * WW + ((s + 1) / HH)) : (LL - 2 - s);
      size_t tn = ((size_t)(b * KD + k)) * LL + pn;
      __builtin_prefetch(&delta_buf[tn * DI + d], 0, 1);
      __builtin_prefetch(&xconv_t[((size_t)b * LL + pn) * DI + d], 0, 1);
    }

    float dt = delta_buf[tok * DI + d];
    float u  = xconv_t[((size_t)b * LL + p) * DI + d];
    float du = dt * u;
    float y = 0.0f;
#pragma unroll
    for (int n = 0; n < NS; ++n) {
      h[n] = __expf(dt * Ad[n]) * h[n] + du * sh[n];
      y = fmaf(h[n], sh[NS + n], y);
    }
    ybuf[tok * DI + d] = fmaf(u, Dsv, y);
    __syncthreads();
  }
}

// ---- Kernel 6: merge three directions and gate with silu(z) ----
__global__ void k_merge(const float* __restrict__ ybuf, const float* __restrict__ zbuf,
                        float* __restrict__ ymul) {
  int idx = blockIdx.x * blockDim.x + threadIdx.x;
  if (idx >= TOK * DI) return;
  int d = idx % DI; int t = idx / DI;
  int b = t / LL, p = t % LL;
  size_t base = ((size_t)b * KD * LL + p) * DI + d;     // k stride = LL*DI
  float y = ybuf[base] + ybuf[base + (size_t)LL * DI] + ybuf[base + 2 * (size_t)LL * DI];
  ymul[idx] = y * zbuf[idx];
}

// ---- Kernel 7: out_proj GEMM (T x 192) @ (192 x 96)^T. One wave = 1 Mtile x all 6 Ntiles. ----
__global__ void k_outproj(const float* __restrict__ ymul, const float* __restrict__ opw,
                          float* __restrict__ out) {
  int lane = threadIdx.x, half = lane >> 4, ln = lane & 15;
  int mt = blockIdx.x;
  const float* arow = ymul + (size_t)(mt * 16 + ln) * DI;
  const float* bbase = opw + (size_t)ln * DI;
  v8f acc[6] = {};
#pragma unroll
  for (int kb = 0; kb < DI; kb += 32) {
    v16bf a = fragA_f32(arow, half, kb);
#pragma unroll
    for (int i = 0; i < 6; ++i) {
      v16bf b = fragB_f32(bbase + (size_t)i * 16 * DI, half, kb);
      acc[i] = __builtin_amdgcn_wmma_f32_16x16x32_bf16(false, a, false, b, (short)0, acc[i], false, false);
    }
  }
#pragma unroll
  for (int i = 0; i < 6; ++i) {
#pragma unroll
    for (int r = 0; r < 8; ++r) {
      int t = mt * 16 + half * 8 + r;
      out[(size_t)t * DM + i * 16 + ln] = acc[i][r];
    }
  }
}

extern "C" void kernel_launch(void* const* d_in, const int* in_sizes, int n_in,
                              void* d_out, int out_size, void* d_ws, size_t ws_size,
                              hipStream_t stream) {
  const float* x    = (const float*)d_in[0];
  // d_in[1] = illum: accepted but unused by the reference
  const float* inpw = (const float*)d_in[2];
  const float* cw   = (const float*)d_in[3];
  const float* cb   = (const float*)d_in[4];
  const float* xpw  = (const float*)d_in[5];
  const float* dtw  = (const float*)d_in[6];
  const float* dtb  = (const float*)d_in[7];
  const float* Alog = (const float*)d_in[8];
  const float* DsP  = (const float*)d_in[9];
  const float* opw  = (const float*)d_in[10];
  float* out = (float*)d_out;

  float* ws      = (float*)d_ws;
  float* xcbuf   = ws;                                         // B*DI*L
  float* zbuf    = xcbuf   + (size_t)BATCH * DI * LL;          // T*DI
  float* xconv   = zbuf    + (size_t)TOK * DI;                 // T*DI
  float* xdbl    = xconv   + (size_t)TOK * DI;                 // B*K*L*38
  float* dtspad  = xdbl    + (size_t)BATCH * KD * LL * CC;     // B*K*L*32
  float* delta   = dtspad  + (size_t)BATCH * KD * LL * DTPAD;  // B*K*L*DI
  float* ybuf    = delta   + (size_t)BATCH * KD * LL * DI;     // B*K*L*DI
  float* ymul    = ybuf    + (size_t)BATCH * KD * LL * DI;     // T*DI
  float* xpw_pad = ymul    + (size_t)TOK * DI;                 // K*48*192
  float* dtw_pad = xpw_pad + (size_t)KD * CPAD * DI;           // K*192*32

  k_pack_xpw<<<(KD * CPAD * DI + 255) / 256, 256, 0, stream>>>(xpw, xpw_pad);
  k_pack_dtw<<<(KD * DI * DTPAD + 255) / 256, 256, 0, stream>>>(dtw, dtw_pad);

  k_inproj<<<dim3(TOK / 16, (2 * DI) / 64), 32, 0, stream>>>(x, inpw, xcbuf, zbuf);
  k_conv<<<(BATCH * DI * LL + 255) / 256, 256, 0, stream>>>(xcbuf, cw, cb, xconv);
  k_xproj<<<dim3(TOK / 16, KD), 32, 0, stream>>>(xconv, xpw_pad, xdbl, dtspad);
  k_dtproj<<<dim3(TOK / 16, 3, KD), 32, 0, stream>>>(dtspad, dtw_pad, dtb, delta);
  k_scan<<<dim3(BATCH * KD), DI, 0, stream>>>(xconv, xdbl, delta, Alog, DsP, ybuf);
  k_merge<<<(TOK * DI + 255) / 256, 256, 0, stream>>>(ybuf, zbuf, ymul);
  k_outproj<<<dim3(TOK / 16, 1), 32, 0, stream>>>(ymul, opw, out);
}
